// NeuralSplineFlow3D_1194000908368
// MI455X (gfx1250) — matrix operand order repaired and checked by reading, not compiled, not run on hardware
//
#include <hip/hip_runtime.h>
#include <hip/hip_bf16.h>

typedef __attribute__((ext_vector_type(16))) _Float16 v16h;
typedef __attribute__((ext_vector_type(8)))  _Float16 v8h;
typedef __attribute__((ext_vector_type(8)))  float    v8f;
typedef __attribute__((ext_vector_type(4)))  int      i32x4;

#define NPTS   1048576
#define NTR    8
#define HID    64
#define NBIN   12
#define TBND   5.0f
#define MINSZ  0.001f
#define OUTD   74
#define OUTDP  80
#define DCONSTF 0.5397416f            /* log(exp(1-0.001)-1) */
#define LOGZF  -2.7568156f            /* -1.5*log(2*pi)      */
#define SZSCALE 0.988f                /* 1 - MINSZ*NBIN      */

#define WAVES   8
#define PTS_PER_WAVE 16
#define PTS_PER_BLOCK (WAVES * PTS_PER_WAVE)   /* 128 */
#define THREADS (WAVES * 32)                   /* 256 */

#define WBLK_HALVES (NTR * 4 * HID * HID)      /* 131072 */
#define WOUT_HALVES (NTR * OUTDP * HID)        /* 40960  */

/* LDS partition (bytes, all 16B aligned), total 60224:
   s_wl   @     0 : 64*64  f16  current residual-layer weights (row-major W[n][k])
   s_wout @  8192 : 80*64  f16  output weights (rows 74..79 zeroed)
   s_bblk @ 18432 : 4*64   f32
   s_bout @ 19456 : 80     f32
   s_win  @ 19776 : 64     f32
   s_bin  @ 20032 : 64     f32
   s_a    @ 20288 : 8*16*64  f16  per-wave A-fragment staging
   s_par  @ 36672 : 8*16*80  f16  per-wave spline params
   s_z    @ 57152 : 8*16*4   f32  per-wave point state
   s_lad  @ 59200 : 8*32     f32  per-lane lad reduce                         */

typedef __attribute__((address_space(1))) i32x4* g128_t;   /* global int4* */
typedef __attribute__((address_space(3))) i32x4* l128_t;   /* LDS    int4* */

__device__ __forceinline__ void wait_dscnt0() {
  asm volatile("s_wait_dscnt 0" ::: "memory");
}

/* Async DMA of one 16B chunk global(f16)->LDS; ASYNCcnt-tracked on gfx1250. */
__device__ __forceinline__ void async_cp16(const _Float16* g, _Float16* l) {
#if __has_builtin(__builtin_amdgcn_global_load_async_to_lds_b128)
  __builtin_amdgcn_global_load_async_to_lds_b128((g128_t)g, (l128_t)l, 0, 0);
#else
  *(v8h*)l = *(const v8h*)g;
#endif
}
__device__ __forceinline__ void wait_async0() {
#if __has_builtin(__builtin_amdgcn_global_load_async_to_lds_b128)
  asm volatile("s_wait_asynccnt 0" ::: "memory");
#endif
}

__device__ __forceinline__ float softplusf(float v) {
  return (v > 20.f) ? v : __logf(1.f + __expf(v));
}

template<int TILES>
__device__ __forceinline__ void wave_gemm(const _Float16* __restrict__ sA,
                                          const _Float16* __restrict__ sW,
                                          v8f (&acc)[TILES], int lane) {
  const int m     = lane & 15;
  const int base0 = (lane >> 4) << 3;   /* A: K half-group 0 or 8   */
  const int kbB   = (lane >> 4) << 3;   /* B: K half-group 0 or 8   */
  const int n     = lane & 15;
  #pragma unroll
  for (int kb = 0; kb < HID; kb += 32) {
    v8h alo = *(const v8h*)&sA[m * HID + kb + base0];
    v8h ahi = *(const v8h*)&sA[m * HID + kb + base0 + 16];
    v16h a = __builtin_shufflevector(alo, ahi, 0,1,2,3,4,5,6,7,8,9,10,11,12,13,14,15);
    #pragma unroll
    for (int t = 0; t < TILES; ++t) {
      const _Float16* wr = &sW[(t * 16 + n) * HID + kb + kbB];
      v8h blo = *(const v8h*)wr;
      v8h bhi = *(const v8h*)(wr + 16);
      v16h b = __builtin_shufflevector(blo, bhi, 0,1,2,3,4,5,6,7,8,9,10,11,12,13,14,15);
      acc[t] = __builtin_amdgcn_wmma_f32_16x16x32_f16(false, a, false, b,
                                                      (short)0, acc[t], false, false);
    }
  }
}

__device__ __forceinline__ void stage_frag(const v8f (&c)[4], _Float16* sA,
                                           int lane, bool dorelu) {
  const int n   = lane & 15;
  const int hi8 = (lane >> 4) << 3;
  #pragma unroll
  for (int t = 0; t < 4; ++t)
    #pragma unroll
    for (int r = 0; r < 8; ++r) {
      float v = c[t][r];
      if (dorelu) v = fmaxf(v, 0.f);
      sA[(r + hi8) * HID + t * 16 + n] = (_Float16)v;
    }
  wait_dscnt0();
}

/* ---- prep: one-time f32 -> f16 weight conversion into workspace ---- */
__global__ void __launch_bounds__(THREADS)
convert_weights(const float* __restrict__ Wblk, const float* __restrict__ Wout,
                _Float16* __restrict__ wsblk, _Float16* __restrict__ wsout) {
  const int idx = blockIdx.x * blockDim.x + threadIdx.x;
  if (idx < WBLK_HALVES) wsblk[idx] = (_Float16)Wblk[idx];
  if (idx < WOUT_HALVES) {
    const int i = idx / (OUTDP * HID);
    const int r = idx - i * (OUTDP * HID);
    const int j = r >> 6, k = r & 63;
    wsout[idx] = (_Float16)((j < OUTD) ? Wout[((size_t)i * OUTD + j) * HID + k] : 0.f);
  }
}

__global__ void __launch_bounds__(THREADS)
nsf3d_kernel(const float* __restrict__ x,
             const float* __restrict__ Win,
             const float* __restrict__ binp,
             const float* __restrict__ bblk,
             const float* __restrict__ bout,
             const _Float16* __restrict__ wsblk,
             const _Float16* __restrict__ wsout,
             float* __restrict__ out) {
  extern __shared__ char smem[];
  _Float16* s_wl   = (_Float16*)(smem);
  _Float16* s_wout = (_Float16*)(smem + 8192);
  float*    s_bblk = (float*)(smem + 18432);
  float*    s_bout = (float*)(smem + 19456);
  float*    s_win  = (float*)(smem + 19776);
  float*    s_bin  = (float*)(smem + 20032);
  _Float16* s_a    = (_Float16*)(smem + 20288);
  _Float16* s_par  = (_Float16*)(smem + 36672);
  float*    s_z    = (float*)(smem + 57152);
  float*    s_lad  = (float*)(smem + 59200);

  const int tid  = threadIdx.x;
  const int lane = tid & 31;
  const int wv   = tid >> 5;
  const int p    = lane & 15;      /* point within wave tile        */
  const int d    = lane >> 4;      /* transformed dim for spline    */
  const int n    = p;              /* fragment column               */
  const int hi8  = d << 3;

  _Float16* s_aw = s_a + wv * (PTS_PER_WAVE * HID);
  _Float16* s_pw = s_par + wv * (PTS_PER_WAVE * OUTDP);
  float*    s_zw = s_z + wv * (PTS_PER_WAVE * 4);

  const int pbase = blockIdx.x * PTS_PER_BLOCK;

  /* load this block's 128 points (x row-major N x 3) */
  for (int idx = tid; idx < PTS_PER_BLOCK * 3; idx += THREADS) {
    int pp = idx / 3, dim = idx - pp * 3;
    s_z[(pp >> 4) * 64 + (pp & 15) * 4 + dim] = x[(size_t)pbase * 3 + idx];
  }

  float ladAcc = 0.f;

  for (int i = 0; i < NTR; ++i) {
    __syncthreads();
    /* async-stage f16 W_out + residual layer-0 weights straight into LDS */
    {
      const _Float16* gwo = wsout + (size_t)i * (OUTDP * HID);
      for (int c = tid; c < (OUTDP * HID) / 8; c += THREADS)
        async_cp16(gwo + c * 8, s_wout + c * 8);
      const _Float16* gwl = wsblk + (size_t)(i * 4) * (HID * HID);
      for (int c = tid; c < (HID * HID) / 8; c += THREADS)
        async_cp16(gwl + c * 8, s_wl + c * 8);
    }
    if (tid < 4 * HID) s_bblk[tid] = bblk[i * 4 * HID + tid];
    if (tid < OUTDP)   s_bout[tid] = (tid < OUTD) ? bout[i * OUTD + tid] : 0.f;
    if (tid < HID) { s_win[tid] = Win[i * HID + tid]; s_bin[tid] = binp[i * HID + tid]; }
    wait_async0();
    __syncthreads();

    /* z = z[:, ::-1] : swap dim0 <-> dim2 (per-wave) */
    if (lane < PTS_PER_WAVE) {
      float a0 = s_zw[lane * 4 + 0];
      float a2 = s_zw[lane * 4 + 2];
      s_zw[lane * 4 + 0] = a2;
      s_zw[lane * 4 + 2] = a0;
    }
    wait_dscnt0();

    /* h = ident * W_in + b_in, directly in C/D fragment layout */
    float idv[8];
    #pragma unroll
    for (int r = 0; r < 8; ++r) idv[r] = s_zw[(r + hi8) * 4 + 0];
    v8f hc[4], tc[4];
    #pragma unroll
    for (int t = 0; t < 4; ++t) {
      float w0 = s_win[t * 16 + n], b0 = s_bin[t * 16 + n];
      #pragma unroll
      for (int r = 0; r < 8; ++r) hc[t][r] = fmaf(idv[r], w0, b0);
    }

    /* 2 residual blocks = 4 WMMA GEMM layers */
    #pragma unroll
    for (int l = 0; l < 4; ++l) {
      if (l > 0) {
        __syncthreads();
        const _Float16* gwl = wsblk + (size_t)(i * 4 + l) * (HID * HID);
        for (int c = tid; c < (HID * HID) / 8; c += THREADS)
          async_cp16(gwl + c * 8, s_wl + c * 8);
        wait_async0();
        __syncthreads();
      }
      if ((l & 1) == 0) stage_frag(hc, s_aw, lane, true);
      else              stage_frag(tc, s_aw, lane, true);
      v8f acc[4];
      #pragma unroll
      for (int t = 0; t < 4; ++t) {
        float bv = s_bblk[l * HID + t * 16 + n];
        #pragma unroll
        for (int r = 0; r < 8; ++r) acc[t][r] = bv;
      }
      wave_gemm<4>(s_aw, s_wl, acc, lane);
      if ((l & 1) == 0) {
        #pragma unroll
        for (int t = 0; t < 4; ++t) tc[t] = acc[t];
      } else {
        #pragma unroll
        for (int t = 0; t < 4; ++t)
          #pragma unroll
          for (int r = 0; r < 8; ++r) hc[t][r] += acc[t][r];
      }
    }

    /* params = h @ W_out^T + b_out (no relu on h) */
    stage_frag(hc, s_aw, lane, false);
    v8f pacc[5];
    #pragma unroll
    for (int t = 0; t < 5; ++t) {
      float bv = s_bout[t * 16 + n];
      #pragma unroll
      for (int r = 0; r < 8; ++r) pacc[t][r] = bv;
    }
    wave_gemm<5>(s_aw, s_wout, pacc, lane);
    #pragma unroll
    for (int t = 0; t < 5; ++t)
      #pragma unroll
      for (int r = 0; r < 8; ++r)
        s_pw[(r + hi8) * OUTDP + t * 16 + n] = (_Float16)pacc[t][r];
    wait_dscnt0();

    /* RQ spline: lane handles (point p, dim d); params[p][d*37 + k] */
    const _Float16* pq = &s_pw[p * OUTDP + d * 37];
    float xin = s_zw[p * 4 + 1 + d];
    bool inside = (xin >= -TBND) && (xin <= TBND);
    float xc = fminf(fmaxf(xin, -TBND), TBND);

    float mw = -1e30f, mh = -1e30f;
    #pragma unroll
    for (int k = 0; k < NBIN; ++k) {
      mw = fmaxf(mw, (float)pq[k]);
      mh = fmaxf(mh, (float)pq[NBIN + k]);
    }
    float sw = 0.f, sh = 0.f;
    #pragma unroll
    for (int k = 0; k < NBIN; ++k) {
      sw += __expf((float)pq[k] - mw);
      sh += __expf((float)pq[NBIN + k] - mh);
    }
    float accw = 0.f, acch = 0.f, leftw = -TBND, lefth = -TBND;
    float in_cw = -TBND, in_w = 1.f, in_ch = -TBND, in_h = 1.f;
    int bin = 0;
    #pragma unroll
    for (int k = 0; k < NBIN; ++k) {
      accw += MINSZ + SZSCALE * __expf((float)pq[k] - mw) / sw;
      acch += MINSZ + SZSCALE * __expf((float)pq[NBIN + k] - mh) / sh;
      float rightw = (k == NBIN - 1) ? TBND : fmaf(2.f * TBND, accw, -TBND);
      float righth = (k == NBIN - 1) ? TBND : fmaf(2.f * TBND, acch, -TBND);
      if (xc >= leftw) {
        bin = k;
        in_cw = leftw; in_w = rightw - leftw;
        in_ch = lefth; in_h = righth - lefth;
      }
      leftw = rightw; lefth = righth;
    }
    float udk  = (bin == 0) ? DCONSTF : (float)pq[2 * NBIN + bin - 1];
    float udk1 = (float)pq[2 * NBIN + bin];
    float dk  = MINSZ + softplusf(udk);
    float dk1 = MINSZ + softplusf(udk1);
    float delta = in_h / in_w;
    float theta = (xc - in_cw) / in_w;
    float t1m = theta * (1.f - theta);
    float num = in_h * (delta * theta * theta + dk * t1m);
    float den = delta + (dk + dk1 - 2.f * delta) * t1m;
    float y = in_ch + num / den;
    float omt = 1.f - theta;
    float dnum = delta * delta * (dk1 * theta * theta + 2.f * delta * t1m + dk * omt * omt);
    float lad = __logf(dnum) - 2.f * __logf(den);
    y   = inside ? y : xin;
    lad = inside ? lad : 0.f;
    ladAcc += lad;
    s_zw[p * 4 + 1 + d] = y;
    wait_dscnt0();
  }

  /* final log-density: -0.5*||z||^2 + LOGZ + sum(lad over both dims) */
  s_lad[wv * 32 + lane] = ladAcc;
  wait_dscnt0();
  if (lane < PTS_PER_WAVE) {
    float z0 = s_zw[lane * 4 + 0];
    float z1 = s_zw[lane * 4 + 1];
    float z2 = s_zw[lane * 4 + 2];
    float lt = s_lad[wv * 32 + lane] + s_lad[wv * 32 + lane + 16];
    out[pbase + wv * PTS_PER_WAVE + lane] =
        -0.5f * (z0 * z0 + z1 * z1 + z2 * z2) + LOGZF + lt;
  }
}

extern "C" void kernel_launch(void* const* d_in, const int* in_sizes, int n_in,
                              void* d_out, int out_size, void* d_ws, size_t ws_size,
                              hipStream_t stream) {
  const float* x    = (const float*)d_in[0];
  const float* Win  = (const float*)d_in[1];
  const float* binp = (const float*)d_in[2];
  const float* Wblk = (const float*)d_in[3];
  const float* bblk = (const float*)d_in[4];
  const float* Wout = (const float*)d_in[5];
  const float* bout = (const float*)d_in[6];
  float* out = (float*)d_out;

  _Float16* wsblk = (_Float16*)d_ws;                /* 131072 halves */
  _Float16* wsout = wsblk + WBLK_HALVES;            /* 40960 halves  */

  const int cblocks = (WBLK_HALVES + THREADS - 1) / THREADS;  /* 512 */
  convert_weights<<<cblocks, THREADS, 0, stream>>>(Wblk, Wout, wsblk, wsout);

  const int nblocks = NPTS / PTS_PER_BLOCK;   /* 8192 */
  const size_t lds_bytes = 60224;
  nsf3d_kernel<<<nblocks, THREADS, lds_bytes, stream>>>(x, Win, binp, bblk, bout,
                                                        wsblk, wsout, out);
  (void)in_sizes; (void)n_in; (void)out_size; (void)ws_size;
}
